// DecoderWrapper_74088185856816
// MI455X (gfx1250) — matrix operand order repaired and checked by reading, not compiled, not run on hardware
//
#include <hip/hip_runtime.h>
#include <hip/hip_bf16.h>
#include <cstdint>

// ---------------------------------------------------------------------------
// T5/FiD cross-attention for MI455X (gfx1250, wave32, WMMA).
// Memory-bound (~420MB HBM round trip ~18us @23.3TB/s vs 139 GFLOP bf16 WMMA).
//  - all GEMMs: v_wmma_f32_16x16x32_bf16 (f32 accumulate)
//  - K/V intermediates bf16; V stored pre-permuted in WMMA-fragment order
//  - K/V chunk staging via global_load_async_to_lds_b128 (+ s_wait_asynccnt)
//  - all WMMA operand LDS loads and all K/V global stores are 16B (b128)
//  - flash attention split 5-way along K per (b,h) for occupancy (640 blocks),
//    merged with log-sum-exp rescale; context sums partition cleanly by split
// ---------------------------------------------------------------------------

typedef __attribute__((ext_vector_type(16))) __bf16 v16bf;
typedef __attribute__((ext_vector_type(8)))  __bf16 v8bf;
typedef __attribute__((ext_vector_type(4)))  __bf16 v4bf;
typedef __attribute__((ext_vector_type(8)))  float  v8f;

#define BSZN 8
#define QL   32
#define KL   4000
#define DM   1024
#define NH   16
#define DH   64
#define NCTX 20
#define PLEN 200
#define CH   160      // attention k-chunk (10 tiles of 16)
#define NCK  25       // total chunks per (b,h)
#define NSP  5        // K-splits per (b,h): 800 keys = 5 chunks = 4 contexts
#define KST  72       // padded bf16 row stride (16B-aligned runs, bank-safe)
#define AST  40       // padded bf16 row stride for A blocks in proj_kv
#define SST  164      // padded f32 row stride for raw scores
#define PST  168      // padded bf16 row stride for probabilities
#define EVST 136      // padded row stride for transposed V epilogue block

__device__ __forceinline__ v16bf cat8(v8bf lo, v8bf hi) {
  return __builtin_shufflevector(lo, hi, 0, 1, 2, 3, 4, 5, 6, 7, 8, 9, 10, 11, 12, 13, 14, 15);
}
// A-operand fragment from row-major bf16 (p = row base + k0): runs at +8h and +16+8h
__device__ __forceinline__ v16bf loadA16(const __bf16* p, int half) {
  return cat8(*(const v8bf*)(p + 8 * half), *(const v8bf*)(p + 16 + 8 * half));
}
// B-operand fragment when the 16 per-lane values are contiguous at p
__device__ __forceinline__ v16bf loadB16(const __bf16* p) {
  return cat8(*(const v8bf*)p, *(const v8bf*)(p + 8));
}
__device__ __forceinline__ v8f wmma_bf16(v16bf a, v16bf b, v8f c) {
  return __builtin_amdgcn_wmma_f32_16x16x32_bf16(false, a, false, b, (short)0, c, false, false);
}

// ---- CDNA5 async global->LDS copy (ASYNCcnt) -------------------------------
__device__ __forceinline__ void async_ld_b128(uint32_t lds_off, const void* g) {
  asm volatile("global_load_async_to_lds_b128 %0, %1, off" :: "v"(lds_off), "v"(g) : "memory");
}
__device__ __forceinline__ void wait_async0() {
  asm volatile("s_wait_asynccnt 0" ::: "memory");
}

// ---------------------------------------------------------------------------
// Kernel 1: Q projection.  X[256,1024] @ Wq -> qbf[b][h][q][d] bf16.
// ---------------------------------------------------------------------------
__global__ __launch_bounds__(128) void k_proj_q(const float* __restrict__ X,
                                                const float* __restrict__ Wq,
                                                __bf16* __restrict__ qbf) {
  const int tid = threadIdx.x;
  const int lane = tid & 31, wave = tid >> 5;
  const int half = lane >> 4, lm = lane & 15;
  const int t = blockIdx.x * 4 + wave;      // 1024 tiles
  const int tm = t >> 6, tn = t & 63;
  const int m0 = tm * 16, n0 = tn * 16;
  v8f c = {};
  for (int k0 = 0; k0 < DM; k0 += 32) {
    const float* ar = X + (size_t)(m0 + lm) * DM + k0 + 8 * half;
    const float4 f0 = *(const float4*)ar;
    const float4 f1 = *(const float4*)(ar + 4);
    const float4 f2 = *(const float4*)(ar + 16);
    const float4 f3 = *(const float4*)(ar + 20);
    v16bf a;
    a[0] = (__bf16)f0.x; a[1] = (__bf16)f0.y; a[2] = (__bf16)f0.z; a[3] = (__bf16)f0.w;
    a[4] = (__bf16)f1.x; a[5] = (__bf16)f1.y; a[6] = (__bf16)f1.z; a[7] = (__bf16)f1.w;
    a[8] = (__bf16)f2.x; a[9] = (__bf16)f2.y; a[10] = (__bf16)f2.z; a[11] = (__bf16)f2.w;
    a[12] = (__bf16)f3.x; a[13] = (__bf16)f3.y; a[14] = (__bf16)f3.z; a[15] = (__bf16)f3.w;
    v16bf b;
    const int bcol = n0 + lm;
#pragma unroll
    for (int i = 0; i < 16; ++i) b[i] = (__bf16)Wq[(size_t)(k0 + (i & 7) + 8 * half + 16 * (i >> 3)) * DM + bcol];
    c = wmma_bf16(a, b, c);
  }
  // tile lies within one batch (m0 multiple of 16 inside 32-row batch blocks)
  const int n = n0 + lm, h = n >> 6, d = n & 63;
  const int bb = m0 >> 5, q0 = (m0 & 31) + 8 * half;
  __bf16* p = qbf + ((size_t)(bb * NH + h) * QL + q0) * DH + d;
#pragma unroll
  for (int r = 0; r < 8; ++r) p[r * DH] = (__bf16)c[r];
}

// ---------------------------------------------------------------------------
// Kernel 2: fused K/V projection, 128x128 blocked; LDS-staged epilogue so all
// global K/V stores are coalesced b128. K -> [b][h][k][d]; V -> fragment order.
// ---------------------------------------------------------------------------
__global__ __launch_bounds__(256) void k_proj_kv(const float* __restrict__ KV,
                                                 const float* __restrict__ Wk,
                                                 const float* __restrict__ Wv,
                                                 __bf16* __restrict__ Kbf,
                                                 __bf16* __restrict__ Vperm) {
  // staging: As 10240B | Bksw 8192B | Bvsw 8192B   (26624B)
  // epilogue: eK 32768B | eV (transposed, padded) 34816B  (67584B)
  __shared__ __align__(16) char smraw[67584];
  __bf16* As   = (__bf16*)smraw;                   // [128][AST]
  __bf16* Bksw = (__bf16*)(smraw + 10240);         // [8*32*16]
  __bf16* Bvsw = (__bf16*)(smraw + 18432);
  __bf16* eK   = (__bf16*)smraw;                   // [128][128] row-major
  __bf16* eV   = (__bf16*)(smraw + 32768);         // [128 cols][EVST] (transposed)

  const int tid = threadIdx.x;
  const int lane = tid & 31, wave = tid >> 5;
  const int half = lane >> 4, lm = lane & 15;
  const int wm = wave >> 1, wn = wave & 1;
  const int mb = blockIdx.y, nb = blockIdx.x;      // 250 x 8

  v8f accK[2][4] = {};
  v8f accV[2][4] = {};

  const int stile = tid >> 5, sln = tid & 31;      // staging role: (tile, lane)
  const int shb = sln >> 4, snn = sln & 15;
  const int scol = nb * 128 + stile * 16 + snn;

  for (int k0 = 0; k0 < DM; k0 += 32) {
#pragma unroll
    for (int j = 0; j < 4; ++j) {                  // A: coalesced float4 -> bf16
      const int elem = (tid + 256 * j) * 4;
      const int r = elem >> 5, cc = elem & 31;
      const float4 f = *(const float4*)&KV[(size_t)(mb * 128 + r) * DM + k0 + cc];
      v4bf v; v[0] = (__bf16)f.x; v[1] = (__bf16)f.y; v[2] = (__bf16)f.z; v[3] = (__bf16)f.w;
      *(v4bf*)&As[r * AST + cc] = v;
    }
    {                                              // Wk/Wv in fragment order
      v8bf k0v, k1v, v0v, v1v;
#pragma unroll
      for (int i = 0; i < 8; ++i) {
        const size_t gi = (size_t)(k0 + i + 8 * shb) * DM + scol;
        k0v[i] = (__bf16)Wk[gi];
        v0v[i] = (__bf16)Wv[gi];
      }
#pragma unroll
      for (int i = 0; i < 8; ++i) {
        const size_t gi = (size_t)(k0 + 16 + i + 8 * shb) * DM + scol;
        k1v[i] = (__bf16)Wk[gi];
        v1v[i] = (__bf16)Wv[gi];
      }
      const int o = (stile * 32 + sln) * 16;
      *(v8bf*)&Bksw[o] = k0v; *(v8bf*)&Bksw[o + 8] = k1v;
      *(v8bf*)&Bvsw[o] = v0v; *(v8bf*)&Bvsw[o + 8] = v1v;
    }
    __syncthreads();

    v16bf a[2];
#pragma unroll
    for (int s = 0; s < 2; ++s)
      a[s] = loadA16(&As[(wm * 32 + s * 16 + lm) * AST], half);
#pragma unroll
    for (int ns = 0; ns < 4; ++ns) {
      const int o = ((wn * 4 + ns) * 32 + lane) * 16;
      const v16bf bk = loadB16(&Bksw[o]);
      const v16bf bv = loadB16(&Bvsw[o]);
#pragma unroll
      for (int s = 0; s < 2; ++s) {
        accK[s][ns] = wmma_bf16(a[s], bk, accK[s][ns]);
        accV[s][ns] = wmma_bf16(a[s], bv, accV[s][ns]);
      }
    }
    __syncthreads();
  }

  // ---- epilogue: stage block in LDS, then coalesced b128 global stores ----
#pragma unroll
  for (int s = 0; s < 2; ++s) {
#pragma unroll
    for (int ns = 0; ns < 4; ++ns) {
      const int r0 = wm * 32 + s * 16;
      const int c0 = wn * 64 + ns * 16;
#pragma unroll
      for (int r = 0; r < 8; ++r) {
        const int rr = r0 + r + 8 * half;
        const int cc = c0 + lm;
        eK[rr * 128 + cc] = (__bf16)accK[s][ns][r];
        eV[cc * EVST + rr] = (__bf16)accV[s][ns][r];   // transposed
      }
    }
  }
  __syncthreads();
#pragma unroll
  for (int j = 0; j < 8; ++j) {                    // K: 8 elems along d
    const int elem = (tid + 256 * j) * 8;
    const int r = elem >> 7, c = elem & 127;
    const int m = mb * 128 + r;
    const int bb = m / KL, kt = m % KL;
    const int n = nb * 128 + c;
    const int h = n >> 6, d = n & 63;
    *(v8bf*)&Kbf[((size_t)(bb * NH + h) * KL + kt) * DH + d] = *(const v8bf*)&eK[elem];
  }
#pragma unroll
  for (int j = 0; j < 8; ++j) {                    // V: 8 elems along kt (i-run)
    const int f = (tid + 256 * j) * 8;
    const int c = f >> 7, r8 = f & 127;            // col, start row (mult of 8)
    const int m = mb * 128 + r8;
    const int bb = m / KL, kt = m % KL;
    const int n = nb * 128 + c;
    const int h = n >> 6, d = n & 63;
    const int bh = bb * NH + h;
    const int ck = kt / CH, rr = kt % CH;
    const int ks = rr >> 5, hk = (rr >> 4) & 1, i0 = rr & 15;   // i0 in {0,8}
    const int dt = d >> 4, ln = hk * 16 + (d & 15);
    *(v8bf*)&Vperm[((size_t)bh * NCK + ck) * (CH * DH) + ((ks * 4 + dt) * 32 + ln) * 16 + i0] =
        *(const v8bf*)&eV[c * EVST + r8];
  }
}

// ---------------------------------------------------------------------------
// Kernel 3: flash attention, grid (NSP, B*H): split sp handles keys
// [sp*800, sp*800+800) = 5 chunks = contexts [sp*4, sp*4+4). Emits
// unnormalized O partials + running (m, l); writes its 4 collected contexts.
// ---------------------------------------------------------------------------
__global__ __launch_bounds__(256) void k_attn(const __bf16* __restrict__ qbf,
                                              const __bf16* __restrict__ Kbf,
                                              const __bf16* __restrict__ Vperm,
                                              const float* __restrict__ mask_add,
                                              const unsigned char* __restrict__ mask_bool,
                                              const float* __restrict__ pos_bias,
                                              float* __restrict__ Opart,
                                              float* __restrict__ Mout,
                                              float* __restrict__ Lout,
                                              float* __restrict__ collected) {
  __shared__ __bf16 Qs[QL * KST];
  __shared__ __bf16 Ks[CH * KST];
  __shared__ __bf16 Vsw[CH * DH];           // fragment order (matches Vperm)
  __shared__ float  Ss[QL * SST];
  __shared__ __bf16 Ps[QL * PST];
  __shared__ float  colsum[QL][4];
  __shared__ float  maskc[CH];
  __shared__ float  maskb[CH];
  __shared__ float  pmax[QL][8];
  __shared__ float  psum[QL][8];
  __shared__ float  mrun[QL], lrun[QL], corr[QL];
  __shared__ float  cnt4[4];

  const int tid = threadIdx.x;
  const int lane = tid & 31, wave = tid >> 5;
  const int half = lane >> 4, lm = lane & 15;
  const int sp = blockIdx.x;                 // 0..4
  const int bh = blockIdx.y;                 // 0..127
  const int b = bh >> 4, h = bh & 15;
  const int qt = wave >> 2, dt = wave & 3;   // this wave's O tile (2x4)

  const uint32_t ksBase = (uint32_t)(uintptr_t)(&Ks[0]);
  const uint32_t vsBase = (uint32_t)(uintptr_t)(&Vsw[0]);

  for (int e = tid; e < QL * 4; e += 256) colsum[e >> 2][e & 3] = 0.0f;
  if (tid < QL) { mrun[tid] = -3.0e38f; lrun[tid] = 0.0f; }
#pragma unroll
  for (int j = 0; j < 2; ++j) {              // Q -> LDS (padded rows)
    const int elem = (tid + 256 * j) * 4;
    const int q = elem >> 6, d = elem & 63;
    *(v4bf*)&Qs[q * KST + d] = *(const v4bf*)(qbf + (size_t)bh * QL * DH + elem);
  }
  __syncthreads();

  v8f oacc = {};

  for (int ck = 0; ck < NSP; ++ck) {
    const int ckg = sp * NSP + ck;           // global chunk 0..24
    const int kb = ckg * CH;
    {
      const __bf16* kg = Kbf + ((size_t)bh * KL + kb) * DH;
      const __bf16* vg = Vperm + ((size_t)bh * NCK + ckg) * (CH * DH);
#pragma unroll
      for (int j = 0; j < 5; ++j) {
        const int g = tid + 256 * j;         // granule = 8 bf16 = 16B
        const int kk = g >> 3, dd = (g & 7) * 8;
        async_ld_b128(ksBase + (uint32_t)(kk * KST + dd) * 2, kg + (size_t)kk * DH + dd);
        async_ld_b128(vsBase + (uint32_t)g * 16, vg + (size_t)g * 8);
      }
      if (tid < CH) {
        maskc[tid] = mask_add[(size_t)b * KL + kb + tid];
        maskb[tid] = mask_bool[(size_t)b * KL + kb + tid] ? 1.0f : 0.0f;
      }
      wait_async0();
    }
    __syncthreads();

    // ---- S = Q @ Kc^T + mask_add + position_bias -> raw scores in Ss ----
    for (int t = wave; t < 20; t += 8) {     // 2 q-tiles x 10 k-tiles
      const int qt2 = t / 10, kt = t % 10;
      v8f s = {};
#pragma unroll
      for (int d0 = 0; d0 < DH; d0 += 32) {
        const v16bf a = loadA16(&Qs[(qt2 * 16 + lm) * KST + d0], half);
        const v16bf bm = loadA16(&Ks[(kt * 16 + lm) * KST + d0], half);
        s = wmma_bf16(a, bm, s);
      }
      const int kcol = kt * 16 + lm;
      const float madd = maskc[kcol];
      const float* pb = pos_bias + ((size_t)h * QL + qt2 * 16 + 8 * half) * KL + kb + kcol;
      float* so = &Ss[(qt2 * 16 + 8 * half) * SST + kcol];
#pragma unroll
      for (int r = 0; r < 8; ++r) so[r * SST] = s[r] + madd + pb[(size_t)r * KL];
    }
    __syncthreads();

    // ---- per-row chunk max ----
    {
      const int row = tid >> 3, seg = tid & 7;
      float pm = -3.0e38f;
#pragma unroll
      for (int c4 = 0; c4 < 5; ++c4) {
        const float4 rv = *(const float4*)&Ss[row * SST + seg * 20 + c4 * 4];
        pm = fmaxf(pm, fmaxf(fmaxf(rv.x, rv.y), fmaxf(rv.z, rv.w)));
      }
      pmax[row][seg] = pm;
    }
    __syncthreads();
    if (tid < QL) {
      float cm = pmax[tid][0];
#pragma unroll
      for (int j = 1; j < 8; ++j) cm = fmaxf(cm, pmax[tid][j]);
      const float nm = fmaxf(mrun[tid], cm);
      corr[tid] = __expf(mrun[tid] - nm);
      mrun[tid] = nm;
    }
    __syncthreads();

    // ---- exponentiate; row sums; raw-score context sum (1 atomic/thread) ----
    {
      const int row = tid >> 3, seg = tid & 7;
      const float nm = mrun[row];
      const int ctxl = (kb + seg * 20) / PLEN - sp * 4;   // run stays in one context
      float acc = 0.0f, csum = 0.0f;
#pragma unroll
      for (int c4 = 0; c4 < 5; ++c4) {
        const int col = seg * 20 + c4 * 4;
        const float4 rv = *(const float4*)&Ss[row * SST + col];
        const float* rp = &rv.x;
        v4bf pk;
#pragma unroll
        for (int jj = 0; jj < 4; ++jj) {
          const float raw = rp[jj];
          csum += raw * maskb[col + jj];
          const float p = __expf(raw - nm);
          pk[jj] = (__bf16)p;
          acc += p;
        }
        *(v4bf*)&Ps[row * PST + col] = pk;
      }
      atomicAdd(&colsum[row][ctxl], csum);
      psum[row][seg] = acc;
    }
    __syncthreads();

    if (tid < QL) {
      float s8 = 0.0f;
#pragma unroll
      for (int j = 0; j < 8; ++j) s8 += psum[tid][j];
      lrun[tid] = lrun[tid] * corr[tid] + s8;
    }

    // ---- O = O*corr + P @ Vc ----
#pragma unroll
    for (int r = 0; r < 8; ++r) oacc[r] *= corr[qt * 16 + r + 8 * half];
#pragma unroll
    for (int ks = 0; ks < CH / 32; ++ks) {
      const v16bf a = loadB16(&Ps[(qt * 16 + lm) * PST + ks * 32 + 16 * half]);
      const v16bf bm = loadB16(&Vsw[((ks * 4 + dt) * 32 + lane) * 16]);
      oacc = wmma_bf16(a, bm, oacc);
    }
    __syncthreads();
  }

  // ---- epilogue: partials + collected for this split's 4 contexts ----
  if (tid < 4) {
    const int cx = sp * 4 + tid;
    float c = 0.0f;
    for (int j = 0; j < PLEN; ++j)
      c += mask_bool[(size_t)b * KL + cx * PLEN + j] ? 1.0f : 0.0f;
    cnt4[tid] = c;
  }
  if (tid < QL) {
    Mout[((size_t)bh * NSP + sp) * QL + tid] = mrun[tid];
    Lout[((size_t)bh * NSP + sp) * QL + tid] = lrun[tid];
  }
  __syncthreads();

  {
    float* op = Opart + (((size_t)bh * NSP + sp) * QL + qt * 16 + 8 * half) * DH + dt * 16 + lm;
#pragma unroll
    for (int r = 0; r < 8; ++r) op[r * DH] = oacc[r];
  }
  for (int e = tid; e < QL * 4; e += 256) {
    const int row = e >> 2, cxl = e & 3;
    collected[((size_t)bh * QL + row) * NCTX + sp * 4 + cxl] = colsum[row][cxl] / cnt4[cxl];
  }
}

// ---------------------------------------------------------------------------
// Kernel 3b: merge the NSP split partials (log-sum-exp rescale) -> ctx f32.
// ---------------------------------------------------------------------------
__global__ __launch_bounds__(256) void k_attn_merge(const float* __restrict__ Opart,
                                                    const float* __restrict__ Mout,
                                                    const float* __restrict__ Lout,
                                                    float* __restrict__ ctxf) {
  __shared__ float w[NSP][QL];
  __shared__ float Linv[QL];
  const int bh = blockIdx.x, b = bh >> 4, h = bh & 15;
  const int tid = threadIdx.x;
  if (tid < QL) {
    float ms[NSP];
    float mx = -3.0e38f;
#pragma unroll
    for (int spp = 0; spp < NSP; ++spp) {
      ms[spp] = Mout[((size_t)bh * NSP + spp) * QL + tid];
      mx = fmaxf(mx, ms[spp]);
    }
    float L = 0.0f;
#pragma unroll
    for (int spp = 0; spp < NSP; ++spp) {
      const float e = __expf(ms[spp] - mx);
      w[spp][tid] = e;
      L += Lout[((size_t)bh * NSP + spp) * QL + tid] * e;
    }
    Linv[tid] = 1.0f / L;
  }
  __syncthreads();
  for (int e = tid; e < QL * DH; e += 256) {
    const int q = e >> 6, d = e & 63;
    float o = 0.0f;
#pragma unroll
    for (int spp = 0; spp < NSP; ++spp)
      o += Opart[(((size_t)bh * NSP + spp) * QL + q) * DH + d] * w[spp][q];
    ctxf[((size_t)b * QL + q) * DM + h * DH + d] = o * Linv[q];
  }
}

// ---------------------------------------------------------------------------
// Kernel 4: output projection.  ctx[256,1024] @ Wo -> out f32.
// ---------------------------------------------------------------------------
__global__ __launch_bounds__(128) void k_proj_o(const float* __restrict__ A,
                                                const float* __restrict__ Wo,
                                                float* __restrict__ out) {
  const int tid = threadIdx.x;
  const int lane = tid & 31, wave = tid >> 5;
  const int half = lane >> 4, lm = lane & 15;
  const int t = blockIdx.x * 4 + wave;
  const int tm = t >> 6, tn = t & 63;
  const int m0 = tm * 16, n0 = tn * 16;
  v8f c = {};
  for (int k0 = 0; k0 < DM; k0 += 32) {
    const float* ar = A + (size_t)(m0 + lm) * DM + k0 + 8 * half;
    const float4 f0 = *(const float4*)ar;
    const float4 f1 = *(const float4*)(ar + 4);
    const float4 f2 = *(const float4*)(ar + 16);
    const float4 f3 = *(const float4*)(ar + 20);
    v16bf a;
    a[0] = (__bf16)f0.x; a[1] = (__bf16)f0.y; a[2] = (__bf16)f0.z; a[3] = (__bf16)f0.w;
    a[4] = (__bf16)f1.x; a[5] = (__bf16)f1.y; a[6] = (__bf16)f1.z; a[7] = (__bf16)f1.w;
    a[8] = (__bf16)f2.x; a[9] = (__bf16)f2.y; a[10] = (__bf16)f2.z; a[11] = (__bf16)f2.w;
    a[12] = (__bf16)f3.x; a[13] = (__bf16)f3.y; a[14] = (__bf16)f3.z; a[15] = (__bf16)f3.w;
    v16bf bmat;
    const int bcol = n0 + lm;
#pragma unroll
    for (int i = 0; i < 16; ++i) bmat[i] = (__bf16)Wo[(size_t)(k0 + (i & 7) + 8 * half + 16 * (i >> 3)) * DM + bcol];
    c = wmma_bf16(a, bmat, c);
  }
  float* p = out + (size_t)(m0 + 8 * half) * DM + n0 + lm;
#pragma unroll
  for (int r = 0; r < 8; ++r) p[(size_t)r * DM] = c[r];
}

// ---------------------------------------------------------------------------
extern "C" void kernel_launch(void* const* d_in, const int* in_sizes, int n_in,
                              void* d_out, int out_size, void* d_ws, size_t ws_size,
                              hipStream_t stream) {
  (void)in_sizes; (void)n_in; (void)out_size; (void)ws_size;
  const float* input     = (const float*)d_in[0];
  const float* kv        = (const float*)d_in[1];
  const float* mask_add  = (const float*)d_in[2];
  const unsigned char* mask_bool = (const unsigned char*)d_in[3];
  const float* pos_bias  = (const float*)d_in[4];
  const float* Wq        = (const float*)d_in[5];
  const float* Wk        = (const float*)d_in[6];
  const float* Wv        = (const float*)d_in[7];
  const float* Wo        = (const float*)d_in[8];

  float* out       = (float*)d_out;                       // [8,32,1024]
  float* collected = out + (size_t)BSZN * QL * DM;        // [8,16,32,20]

  // ws: qbf 1MiB | Kbf 64MiB | Vperm 64MiB | ctx 1MiB | Opart 8MiB | M/L 1MiB
  char* ws = (char*)d_ws;
  __bf16* qbf   = (__bf16*)ws;
  __bf16* Kbf   = (__bf16*)(ws + (size_t)(1u << 20));
  __bf16* Vperm = (__bf16*)(ws + (size_t)(65u << 20));
  float*  ctx   = (float*) (ws + (size_t)(129u << 20));
  float*  Opart = (float*) (ws + (size_t)(130u << 20));   // [bh][sp][32][64]
  float*  Mout  = (float*) (ws + (size_t)(138u << 20));   // [bh][sp][32]
  float*  Lout  = (float*) (ws + (size_t)(138u << 20) + (512u << 10));

  hipLaunchKernelGGL(k_proj_q,  dim3(256),            dim3(128), 0, stream, input, Wq, qbf);
  hipLaunchKernelGGL(k_proj_kv, dim3(8, 250),         dim3(256), 0, stream, kv, Wk, Wv, Kbf, Vperm);
  hipLaunchKernelGGL(k_attn,    dim3(NSP, BSZN * NH), dim3(256), 0, stream,
                     qbf, Kbf, Vperm, mask_add, mask_bool, pos_bias,
                     Opart, Mout, Lout, collected);
  hipLaunchKernelGGL(k_attn_merge, dim3(BSZN * NH),   dim3(256), 0, stream,
                     Opart, Mout, Lout, ctx);
  hipLaunchKernelGGL(k_proj_o,  dim3(256),            dim3(128), 0, stream, ctx, Wo, out);
}
